// OfficialVideoAttn_83872121356831
// MI455X (gfx1250) — compile-verified, matrix operand order checked by reading, not run on hardware
//
#include <hip/hip_runtime.h>
#include <hip/hip_bf16.h>

#define LSEQ   2048
#define DMODEL 1024
#define NHEADS 16
#define HDIM   64

typedef __bf16 bf16_t;
typedef __bf16 v16bf __attribute__((ext_vector_type(16)));
typedef float  v8f   __attribute__((ext_vector_type(8)));

union FragU { v16bf v; uint4 q[2]; };

static __device__ __forceinline__ v8f vz8() {
  v8f z;
#pragma unroll
  for (int i = 0; i < 8; ++i) z[i] = 0.0f;
  return z;
}

// Per-lane 16-bit A/B fragment for v_wmma_*_16x16x32_*:
// lane holds row `row` (A: M-row / B: N-col as rows of W / K / Q / vT),
// elements K = k0+c+[0..7] and k0+c+16+[0..7], c = (lane<16 ? 0 : 8).
static __device__ __forceinline__ v16bf ld_frag(const bf16_t* __restrict__ base,
                                                int row, int rstride, int k0, int lane) {
  const int c = (lane & 16) ? 8 : 0;
  const bf16_t* p = base + (size_t)row * (size_t)rstride + (size_t)(k0 + c);
  FragU f;
  f.q[0] = *reinterpret_cast<const uint4*>(p);
  f.q[1] = *reinterpret_cast<const uint4*>(p + 16);
  return f.v;
}

static __device__ __forceinline__ v16bf ld_frag_lds(const bf16_t* p0, int lane) {
  const int c = (lane & 16) ? 8 : 0;
  const bf16_t* p = p0 + c;
  FragU f;
  f.q[0] = *reinterpret_cast<const uint4*>(p);
  f.q[1] = *reinterpret_cast<const uint4*>(p + 16);
  return f.v;
}

static __device__ __forceinline__ v8f wmma_bf16(v16bf a, v16bf b, v8f c) {
  return __builtin_amdgcn_wmma_f32_16x16x32_bf16(false, a, false, b, (short)0, c, false, false);
}

// ---------------- Kernel 1: f32 -> bf16 conversion ----------------
__global__ void k_cvt_bf16(const float* __restrict__ src, bf16_t* __restrict__ dst, int n) {
  int i = blockIdx.x * blockDim.x + threadIdx.x;
  if (i < n) dst[i] = (bf16_t)src[i];
}

__global__ void k_rope_tab(const float* __restrict__ rope, float* __restrict__ cb,
                           float* __restrict__ sb, int n) {
  int i = blockIdx.x * blockDim.x + threadIdx.x;
  if (i < n) { float a = rope[i]; cb[i] = __cosf(a); sb[i] = __sinf(a); }
}

// ---------------- Kernel 2: QKV projection + RoPE ----------------
__global__ __launch_bounds__(128)
void k_qkv(const bf16_t* __restrict__ xb,
           const bf16_t* __restrict__ wq, const bf16_t* __restrict__ wk,
           const bf16_t* __restrict__ wv,
           const float* __restrict__ cosb, const float* __restrict__ sinb,
           bf16_t* __restrict__ qb, bf16_t* __restrict__ kb, bf16_t* __restrict__ vT) {
  const int lane  = threadIdx.x & 31;
  const int wid   = blockIdx.x * 4 + (threadIdx.x >> 5);
  const int mat   = wid / (NHEADS * (LSEQ / 16));           // 0=Q 1=K 2=V
  const int rem   = wid % (NHEADS * (LSEQ / 16));
  const int head  = rem / (LSEQ / 16);
  const int mt    = rem % (LSEQ / 16);
  const int m0    = mt * 16;
  const int ln    = lane & 15;
  const int half8 = (lane & 16) ? 8 : 0;
  const bf16_t* W = (mat == 0) ? wq : (mat == 1) ? wk : wv;
  const int arow  = m0 + ln;

  v8f acc[4];
#pragma unroll
  for (int nt = 0; nt < 4; ++nt) acc[nt] = vz8();

  for (int k0 = 0; k0 < DMODEL; k0 += 32) {
    __builtin_prefetch(xb + (size_t)arow * DMODEL + k0 + 128, 0, 3);
    v16bf a = ld_frag(xb, arow, DMODEL, k0, lane);
#pragma unroll
    for (int nt = 0; nt < 4; ++nt) {
      v16bf b = ld_frag(W, head * HDIM + nt * 16 + ln, DMODEL, k0, lane);
      acc[nt] = wmma_bf16(a, b, acc[nt]);
    }
  }

  if (mat < 2) {
    const float sc = (mat == 0) ? 0.125f : 1.0f;   // fold 1/sqrt(hd) into Q
    bf16_t* dst = (mat == 0) ? qb : kb;
#pragma unroll
    for (int r = 0; r < 8; ++r) {
      const int l = m0 + r + half8;
      const float* cp = cosb + (size_t)l * HDIM + ln;
      const float* sp = sinb + (size_t)l * HDIM + ln;
      float c0 = cp[0], c1 = cp[16], c2 = cp[32], c3 = cp[48];
      float s0 = sp[0], s1 = sp[16], s2 = sp[32], s3 = sp[48];
      float q0 = acc[0][r], q1 = acc[1][r], q2 = acc[2][r], q3 = acc[3][r];
      // rotate_half: out[j] = q[j]*cos[j] + (j<32 ? -q[j+32] : q[j-32])*sin[j]
      float o0 = (q0 * c0 - q2 * s0) * sc;
      float o1 = (q1 * c1 - q3 * s1) * sc;
      float o2 = (q2 * c2 + q0 * s2) * sc;
      float o3 = (q3 * c3 + q1 * s3) * sc;
      bf16_t* d = dst + (size_t)head * LSEQ * HDIM + (size_t)l * HDIM + ln;
      d[0] = (bf16_t)o0; d[16] = (bf16_t)o1; d[32] = (bf16_t)o2; d[48] = (bf16_t)o3;
    }
  } else {
    // V stored transposed: vT[head][d][l] so P@V B-fragments are row loads
#pragma unroll
    for (int r = 0; r < 8; ++r) {
      const int l = m0 + r + half8;
#pragma unroll
      for (int nt = 0; nt < 4; ++nt) {
        const int j = nt * 16 + ln;
        vT[((size_t)head * HDIM + j) * LSEQ + l] = (bf16_t)acc[nt][r];
      }
    }
  }
}

// ---------------- Kernel 3: flash attention per head ----------------
// Computes S^T = K_tile @ Q^T so the softmax key-dimension lies along
// registers (cheap VALU) instead of lanes (shuffle-heavy).
__global__ __launch_bounds__(128)
void k_attn(const bf16_t* __restrict__ qb, const bf16_t* __restrict__ kb,
            const bf16_t* __restrict__ vT, bf16_t* __restrict__ ao) {
  __shared__ bf16_t ldsP[4 * 16 * 32];
  const int lane  = threadIdx.x & 31;
  const int w     = threadIdx.x >> 5;
  const int head  = blockIdx.y;
  const int m0    = blockIdx.x * 64 + w * 16;
  const int ln    = lane & 15;
  const int half8 = (lane & 16) ? 8 : 0;
  const bf16_t* qh = qb + (size_t)head * LSEQ * HDIM;
  const bf16_t* kh = kb + (size_t)head * LSEQ * HDIM;
  const bf16_t* vh = vT + (size_t)head * HDIM * LSEQ;
  bf16_t* plds = ldsP + w * 16 * 32;

  // Q is the B operand of S^T = K @ Q^T: loop-invariant, loaded once.
  v16bf bq0 = ld_frag(qh, m0 + ln, HDIM, 0, lane);
  v16bf bq1 = ld_frag(qh, m0 + ln, HDIM, 32, lane);

  v8f o[4];                        // O tile 16(query) x 64(d), C layout
#pragma unroll
  for (int nt = 0; nt < 4; ++nt) o[nt] = vz8();
  float mq = -3.0e38f, lq = 0.0f;  // per-lane stats for query = ln

  for (int kt = 0; kt < LSEQ; kt += 32) {
    __builtin_prefetch(kh + (size_t)(kt + 32 + ln) * HDIM, 0, 3);
    // S^T tiles: rows = keys, cols = queries
    v16bf a00 = ld_frag(kh, kt + ln,      HDIM, 0,  lane);
    v16bf a01 = ld_frag(kh, kt + ln,      HDIM, 32, lane);
    v16bf a10 = ld_frag(kh, kt + 16 + ln, HDIM, 0,  lane);
    v16bf a11 = ld_frag(kh, kt + 16 + ln, HDIM, 32, lane);
    v8f s0 = vz8(), s1 = vz8();
    s0 = wmma_bf16(a00, bq0, s0);
    s0 = wmma_bf16(a01, bq1, s0);
    s1 = wmma_bf16(a10, bq0, s1);
    s1 = wmma_bf16(a11, bq1, s1);

    // ---- online softmax over keys (register dimension) ----
    float tmax = fmaxf(s0[0], s1[0]);
#pragma unroll
    for (int r = 1; r < 8; ++r) tmax = fmaxf(tmax, fmaxf(s0[r], s1[r]));
    tmax = fmaxf(tmax, __shfl_xor(tmax, 16));      // combine key halves
    const float nm = fmaxf(mq, tmax);
    const float cf = __expf(mq - nm);
    mq = nm;

    float psum = 0.0f;
#pragma unroll
    for (int r = 0; r < 8; ++r) {
      float p0 = __expf(s0[r] - nm);
      float p1 = __expf(s1[r] - nm);
      psum += p0 + p1;
      // P lands directly in A-fragment row-major order: plds[query*32 + key]
      plds[ln * 32 + r + half8]      = (bf16_t)p0;   // keys kt..kt+15
      plds[ln * 32 + 16 + r + half8] = (bf16_t)p1;   // keys kt+16..kt+31
    }
    psum += __shfl_xor(psum, 16);
    lq = lq * cf + psum;

    // broadcast per-query rescale onto O's row layout (row M = r + half8)
#pragma unroll
    for (int r = 0; r < 8; ++r) {
      const float cfr = __shfl(cf, r + half8);
      o[0][r] *= cfr; o[1][r] *= cfr; o[2][r] *= cfr; o[3][r] *= cfr;
    }
    asm volatile("s_wait_dscnt 0x0" ::: "memory");  // same-wave LDS RAW

    v16bf ap = ld_frag_lds(plds + ln * 32, lane);   // P as A fragment (K=32 keys)
#pragma unroll
    for (int nt = 0; nt < 4; ++nt) {
      v16bf bv = ld_frag(vh, nt * 16 + ln, LSEQ, kt, lane);  // vT rows = V cols
      o[nt] = wmma_bf16(ap, bv, o[nt]);
    }
  }

  const float inv = 1.0f / lq;
#pragma unroll
  for (int r = 0; r < 8; ++r) {
    const float invr = __shfl(inv, r + half8);
    const int l = m0 + r + half8;
    bf16_t* d = ao + (size_t)l * DMODEL + head * HDIM + ln;
    d[0]  = (bf16_t)(o[0][r] * invr);
    d[16] = (bf16_t)(o[1][r] * invr);
    d[32] = (bf16_t)(o[2][r] * invr);
    d[48] = (bf16_t)(o[3][r] * invr);
  }
}

// ---------------- Kernel 4: output projection -> f32 out ----------------
__global__ __launch_bounds__(128)
void k_oproj(const bf16_t* __restrict__ ao, const bf16_t* __restrict__ wo,
             float* __restrict__ out) {
  const int lane  = threadIdx.x & 31;
  const int wid   = blockIdx.x * 4 + (threadIdx.x >> 5);
  const int ns    = wid / (LSEQ / 16);
  const int mt    = wid % (LSEQ / 16);
  const int m0    = mt * 16;
  const int n0    = ns * 64;
  const int ln    = lane & 15;
  const int half8 = (lane & 16) ? 8 : 0;

  v8f acc[4];
#pragma unroll
  for (int nt = 0; nt < 4; ++nt) acc[nt] = vz8();

  for (int k0 = 0; k0 < DMODEL; k0 += 32) {
    __builtin_prefetch(ao + (size_t)(m0 + ln) * DMODEL + k0 + 128, 0, 3);
    v16bf a = ld_frag(ao, m0 + ln, DMODEL, k0, lane);
#pragma unroll
    for (int nt = 0; nt < 4; ++nt) {
      v16bf b = ld_frag(wo, n0 + nt * 16 + ln, DMODEL, k0, lane);
      acc[nt] = wmma_bf16(a, b, acc[nt]);
    }
  }
#pragma unroll
  for (int r = 0; r < 8; ++r) {
    const int l = m0 + r + half8;
    float* d = out + (size_t)l * DMODEL + n0 + ln;
    d[0] = acc[0][r]; d[16] = acc[1][r]; d[32] = acc[2][r]; d[48] = acc[3][r];
  }
}

extern "C" void kernel_launch(void* const* d_in, const int* in_sizes, int n_in,
                              void* d_out, int out_size, void* d_ws, size_t ws_size,
                              hipStream_t stream) {
  (void)in_sizes; (void)n_in; (void)out_size; (void)ws_size;
  const float* x    = (const float*)d_in[0];
  const float* rope = (const float*)d_in[1];
  const float* Wq   = (const float*)d_in[2];
  const float* Wk   = (const float*)d_in[3];
  const float* Wv   = (const float*)d_in[4];
  const float* Wo   = (const float*)d_in[5];

  char* ws = (char*)d_ws;
  size_t off = 0;
  auto take = [&](size_t bytes) { void* p = ws + off; off += (bytes + 255) & ~(size_t)255; return p; };

  bf16_t* xb   = (bf16_t*)take((size_t)LSEQ * DMODEL * 2);
  bf16_t* wqb  = (bf16_t*)take((size_t)DMODEL * DMODEL * 2);
  bf16_t* wkb  = (bf16_t*)take((size_t)DMODEL * DMODEL * 2);
  bf16_t* wvb  = (bf16_t*)take((size_t)DMODEL * DMODEL * 2);
  bf16_t* wob  = (bf16_t*)take((size_t)DMODEL * DMODEL * 2);
  bf16_t* qb   = (bf16_t*)take((size_t)LSEQ * DMODEL * 2);
  bf16_t* kb   = (bf16_t*)take((size_t)LSEQ * DMODEL * 2);
  bf16_t* vT   = (bf16_t*)take((size_t)LSEQ * DMODEL * 2);
  bf16_t* aob  = (bf16_t*)take((size_t)LSEQ * DMODEL * 2);
  float*  cosb = (float*)take((size_t)LSEQ * HDIM * 4);
  float*  sinb = (float*)take((size_t)LSEQ * HDIM * 4);

  const int nx = LSEQ * DMODEL;
  const int nw = DMODEL * DMODEL;
  const int nr = LSEQ * HDIM;
  k_cvt_bf16<<<(nx + 255) / 256, 256, 0, stream>>>(x, xb, nx);
  k_cvt_bf16<<<(nw + 255) / 256, 256, 0, stream>>>(Wq, wqb, nw);
  k_cvt_bf16<<<(nw + 255) / 256, 256, 0, stream>>>(Wk, wkb, nw);
  k_cvt_bf16<<<(nw + 255) / 256, 256, 0, stream>>>(Wv, wvb, nw);
  k_cvt_bf16<<<(nw + 255) / 256, 256, 0, stream>>>(Wo, wob, nw);
  k_rope_tab<<<(nr + 255) / 256, 256, 0, stream>>>(rope, cosb, sinb, nr);

  // 3 mats * 16 heads * 128 M-tiles = 6144 wave-jobs / 4 waves per block
  k_qkv<<<1536, 128, 0, stream>>>(xb, wqb, wkb, wvb, cosb, sinb, qb, kb, vT);
  k_attn<<<dim3(LSEQ / 64, NHEADS), 128, 0, stream>>>(qb, kb, vT, aob);
  // 16 N-strips * 128 M-tiles = 2048 wave-jobs / 4 per block
  k_oproj<<<512, 128, 0, stream>>>(aob, wob, (float*)d_out);
}
